// BasicTransformerBlock_56934086475875
// MI455X (gfx1250) — matrix operand order repaired and checked by reading, not compiled, not run on hardware
//
#include <hip/hip_runtime.h>

typedef unsigned int u32;
typedef unsigned short u16;
typedef __attribute__((ext_vector_type(16))) __bf16 v16bf;
typedef __attribute__((ext_vector_type(8))) float v8f;

constexpr int kB = 2, kS = 2048, kH = 768, kNH = 12, kDH = 64, kFF = 3072;
constexpr int kM = kB * kS;  // 4096 token rows

__device__ __forceinline__ u16 f2bf(float f) {
  u32 u = __float_as_uint(f);
  u32 r = (u + 0x7FFFu + ((u >> 16) & 1u)) >> 16;
  return (u16)r;
}

struct Frag { uint4 q[2]; };
__device__ __forceinline__ v16bf frag_cast(const Frag &f) {
  v16bf v; __builtin_memcpy(&v, &f, sizeof(v)); return v;
}

// ---------------- weight conversion ----------------
__global__ void k_cvt(const float* __restrict__ src, u16* __restrict__ dst, int n) {
  int i = blockIdx.x * blockDim.x + threadIdx.x;
  if (i < n) dst[i] = f2bf(src[i]);
}

__global__ void k_cvt_qkv(const float* __restrict__ Wq, const float* __restrict__ Wk,
                          const float* __restrict__ Wv, u16* __restrict__ dst) {
  int i = blockIdx.x * blockDim.x + threadIdx.x;
  if (i >= kH * 3 * kH) return;
  int k = i / (3 * kH), c = i % (3 * kH);
  int sel = c / kH, j = c % kH;
  const float* W = (sel == 0) ? Wq : ((sel == 1) ? Wk : Wv);
  dst[i] = f2bf(W[k * kH + j]);
}

// ---------------- layernorm (row = 768) -> bf16 ----------------
__global__ __launch_bounds__(256) void k_ln(const float* __restrict__ x,
                                            const float* __restrict__ g,
                                            const float* __restrict__ bta,
                                            u16* __restrict__ out) {
  __shared__ float red[256];
  int row = blockIdx.x, tid = threadIdx.x;
  const float* xr = x + (size_t)row * kH;
  float v0 = xr[tid], v1 = xr[tid + 256], v2 = xr[tid + 512];
  red[tid] = v0 + v1 + v2;
  __syncthreads();
  for (int o = 128; o > 0; o >>= 1) { if (tid < o) red[tid] += red[tid + o]; __syncthreads(); }
  float mu = red[0] * (1.0f / kH);
  __syncthreads();
  float d0 = v0 - mu, d1 = v1 - mu, d2 = v2 - mu;
  red[tid] = d0 * d0 + d1 * d1 + d2 * d2;
  __syncthreads();
  for (int o = 128; o > 0; o >>= 1) { if (tid < o) red[tid] += red[tid + o]; __syncthreads(); }
  float rs = rsqrtf(red[0] * (1.0f / kH) + 1e-6f);
  u16* orow = out + (size_t)row * kH;
  orow[tid]       = f2bf(d0 * rs * g[tid]       + bta[tid]);
  orow[tid + 256] = f2bf(d1 * rs * g[tid + 256] + bta[tid + 256]);
  orow[tid + 512] = f2bf(d2 * rs * g[tid + 512] + bta[tid + 512]);
}

// ---------------- bf16 WMMA GEMM, block tile 128x128, wave tile 32x64 ----------------
// MODE 0: QKV -> scatter Q[B,NH,S,DH](*0.125), Kt[B,NH,DH,S], V[B,NH,S,DH]
// MODE 1: O-proj: out = C + bias + resid (fp32)
// MODE 2: FFN1:   out = gelu(C + bias)   (bf16)
// MODE 3: FFN2:   out = C + bias + resid (fp32)
template <int MODE>
__global__ __launch_bounds__(256) void k_gemm(
    const u16* __restrict__ A, const u16* __restrict__ B, int K, int N,
    const float* __restrict__ bias_q, const float* __restrict__ bias_k,
    const float* __restrict__ bias_v, const float* __restrict__ resid,
    float* __restrict__ outF, u16* __restrict__ outBf,
    u16* __restrict__ outQ, u16* __restrict__ outKt, u16* __restrict__ outV) {
  // A staged as [row][kpair] (natural), B staged swizzled as [col][kpair]
  __shared__ __align__(16) u32 As[128 * 16];   // 8 KB
  __shared__ __align__(16) u32 BsP[128 * 16];  // 8 KB
  uint4* As4 = (uint4*)As;
  uint4* Bs4 = (uint4*)BsP;

  int tid = threadIdx.x;
  int lane = tid & 31, wid = tid >> 5;
  int half = lane >> 4, n16 = lane & 15;
  int waveM = wid & 3, waveN = wid >> 2;       // 4 x 2 waves
  int m0 = blockIdx.y * 128, n0 = blockIdx.x * 128;

  v8f zero8 = {0.f, 0.f, 0.f, 0.f, 0.f, 0.f, 0.f, 0.f};
  v8f acc[2][4] = {{zero8, zero8, zero8, zero8}, {zero8, zero8, zero8, zero8}};

  const u32* A32 = (const u32*)A;
  const u32* B32 = (const u32*)B;
  int Kw = K >> 1, Nw = N >> 1;

  for (int k0 = 0; k0 < K; k0 += 32) {
    if (k0 + 32 < K)
      __builtin_prefetch(&A32[(size_t)(m0 + (tid >> 4)) * Kw + ((k0 + 32) >> 1) + (tid & 15)], 0, 0);
    __syncthreads();
    // A tile: 128 rows x 4 uint4 -> 512 uint4, 2 per thread (b128 load + b128 store)
#pragma unroll
    for (int i = 0; i < 2; ++i) {
      int idx = tid + i * 256;
      int r = idx >> 2, q = idx & 3;
      const uint4* src = (const uint4*)(A32 + (size_t)(m0 + r) * Kw + (k0 >> 1));
      As4[r * 4 + q] = src[q];
    }
    // B tile: 32 rows x 128 cols, repacked to [col][kpair]; 16 rowpairs x 64 colpairs
#pragma unroll
    for (int i = 0; i < 4; ++i) {
      int u = tid + i * 256;
      int rp = u >> 6, cp = u & 63;
      u32 a = B32[(size_t)(k0 + 2 * rp) * Nw + (n0 >> 1) + cp];
      u32 b = B32[(size_t)(k0 + 2 * rp + 1) * Nw + (n0 >> 1) + cp];
      u32 lo = (a & 0xffffu) | (b << 16);
      u32 hi = (a >> 16) | (b & 0xffff0000u);
      BsP[(2 * cp) * 16 + rp] = lo;
      BsP[(2 * cp + 1) * 16 + rp] = hi;
    }
    __syncthreads();

    Frag af[2], bfr[4];
#pragma unroll
    for (int mm = 0; mm < 2; ++mm) {
      int rowb = waveM * 32 + mm * 16 + n16;
      af[mm].q[0] = As4[rowb * 4 + half];
      af[mm].q[1] = As4[rowb * 4 + 2 + half];
    }
#pragma unroll
    for (int nn = 0; nn < 4; ++nn) {
      int coln = waveN * 64 + nn * 16 + n16;
      bfr[nn].q[0] = Bs4[coln * 4 + half];
      bfr[nn].q[1] = Bs4[coln * 4 + 2 + half];
    }
#pragma unroll
    for (int mm = 0; mm < 2; ++mm)
#pragma unroll
      for (int nn = 0; nn < 4; ++nn)
        acc[mm][nn] = __builtin_amdgcn_wmma_f32_16x16x32_bf16(
            false, frag_cast(af[mm]), false, frag_cast(bfr[nn]),
            (short)0, acc[mm][nn], false, false);
  }

#pragma unroll
  for (int mm = 0; mm < 2; ++mm)
#pragma unroll
    for (int nn = 0; nn < 4; ++nn)
#pragma unroll
      for (int r = 0; r < 8; ++r) {
        int gm = m0 + waveM * 32 + mm * 16 + r + half * 8;
        int gn = n0 + waveN * 64 + nn * 16 + n16;
        float c = acc[mm][nn][r];
        if (MODE == 0) {
          int sel = gn / kH, hc = gn % kH;
          int head = hc / kDH, dh = hc % kDH;
          int b = gm >> 11, s = gm & (kS - 1);
          const float* bias = (sel == 0) ? bias_q : ((sel == 1) ? bias_k : bias_v);
          float val = c + bias[hc];
          int bh = b * kNH + head;
          if (sel == 0)
            outQ[((size_t)(bh * kS + s)) * kDH + dh] = f2bf(val * 0.125f);
          else if (sel == 1)
            outKt[((size_t)(bh * kDH + dh)) * kS + s] = f2bf(val);
          else
            outV[((size_t)(bh * kS + s)) * kDH + dh] = f2bf(val);
        } else if (MODE == 1 || MODE == 3) {
          float val = c + bias_q[gn] + resid[(size_t)gm * N + gn];
          outF[(size_t)gm * N + gn] = val;
        } else {  // MODE 2: exact-erf GELU
          float xx = c + bias_q[gn];
          float gl = 0.5f * xx * (1.0f + erff(xx * 0.70710678118654752f));
          outBf[(size_t)gm * N + gn] = f2bf(gl);
        }
      }
}

// ---------------- flash attention: block = (b,h,128 queries), wave = 16 queries
// 64 keys per iteration: 16 WMMAs per softmax-reduction round
__global__ __launch_bounds__(256) void k_attn(const u16* __restrict__ Q,
                                              const u16* __restrict__ Kt,
                                              const u16* __restrict__ V,
                                              u16* __restrict__ ctx) {
  // ktP: [key][dh-pair]  64 x 32 dwords (8KB); vtP: [dh][key-pair] 64 x 32 dwords (8KB)
  __shared__ __align__(16) u32 ktP[64 * 32];
  __shared__ __align__(16) u32 vtP[64 * 32];
  __shared__ __align__(16) u32 pwm[8 * 512];   // per-wave 16x64 bf16 P scratch (16KB)
  uint4* kt4 = (uint4*)ktP;
  uint4* vt4 = (uint4*)vtP;

  int tid = threadIdx.x, lane = tid & 31, wid = tid >> 5;
  int half = lane >> 4, n16 = lane & 15;
  int bh = blockIdx.x >> 4;   // (b*NH + h)
  int qb = blockIdx.x & 15;
  int q0 = qb * 128 + wid * 16;
  int b = bh / kNH, h = bh % kNH;

  const u32* Q32 = (const u32*)Q;
  const u32* Kt32 = (const u32*)Kt;
  const u32* V32 = (const u32*)V;
  u16* pw16 = (u16*)(pwm + wid * 512);
  uint4* pw4 = (uint4*)(pwm + wid * 512);

  // Q fragments (16x64 = two 32-wide k-steps); Q pre-scaled by 1/sqrt(DH)
  Frag aq[2];
  {
    const uint4* qrow = (const uint4*)(Q32 + (size_t)(bh * kS + q0 + n16) * (kDH >> 1));
#pragma unroll
    for (int ks = 0; ks < 2; ++ks) {
      aq[ks].q[0] = qrow[ks * 4 + half];
      aq[ks].q[1] = qrow[ks * 4 + 2 + half];
    }
  }

  v8f zero8 = {0.f, 0.f, 0.f, 0.f, 0.f, 0.f, 0.f, 0.f};
  float mrow[8], lrow[8];
  v8f accO[4] = {zero8, zero8, zero8, zero8};
#pragma unroll
  for (int r = 0; r < 8; ++r) { mrow[r] = -3.0e38f; lrow[r] = 0.f; }

  for (int kc = 0; kc < kS; kc += 64) {
    __syncthreads();
    // K tile repack: 32 dh-pairs x 32 key-pairs -> ktP[key][dhp]
#pragma unroll
    for (int i = 0; i < 4; ++i) {
      int u = tid + i * 256;
      int dp = u >> 5, cp = u & 31;
      u32 a = Kt32[(size_t)(bh * kDH + 2 * dp) * (kS >> 1) + (kc >> 1) + cp];
      u32 b2 = Kt32[(size_t)(bh * kDH + 2 * dp + 1) * (kS >> 1) + (kc >> 1) + cp];
      u32 lo = (a & 0xffffu) | (b2 << 16);
      u32 hi = (a >> 16) | (b2 & 0xffff0000u);
      ktP[(2 * cp) * 32 + dp] = lo;
      ktP[(2 * cp + 1) * 32 + dp] = hi;
    }
    // V tile repack: 32 key-pairs x 32 dh-pairs -> vtP[dh][keyp]
#pragma unroll
    for (int i = 0; i < 4; ++i) {
      int u = tid + i * 256;
      int rp = u >> 5, cp = u & 31;
      u32 a = V32[(size_t)(bh * kS + kc + 2 * rp) * (kDH >> 1) + cp];
      u32 b2 = V32[(size_t)(bh * kS + kc + 2 * rp + 1) * (kDH >> 1) + cp];
      u32 lo = (a & 0xffffu) | (b2 << 16);
      u32 hi = (a >> 16) | (b2 & 0xffff0000u);
      vtP[(2 * cp) * 32 + rp] = lo;
      vtP[(2 * cp + 1) * 32 + rp] = hi;
    }
    __syncthreads();

    // scores 16x64 = four 16x16 tiles, each K=64 via 2 chained WMMAs
    v8f sc[4];
#pragma unroll
    for (int ct = 0; ct < 4; ++ct) {
      v8f c = zero8;
      int col = ct * 16 + n16;
#pragma unroll
      for (int ks = 0; ks < 2; ++ks) {
        Frag bk;
        bk.q[0] = kt4[col * 8 + ks * 4 + half];
        bk.q[1] = kt4[col * 8 + ks * 4 + 2 + half];
        c = __builtin_amdgcn_wmma_f32_16x16x32_bf16(
            false, frag_cast(aq[ks]), false, frag_cast(bk), (short)0, c, false, false);
      }
      sc[ct] = c;
    }

    // online softmax (row stats replicated across each 16-lane half)
#pragma unroll
    for (int r = 0; r < 8; ++r) {
      float v0 = sc[0][r], v1 = sc[1][r], v2 = sc[2][r], v3 = sc[3][r];
      float mx = fmaxf(fmaxf(v0, v1), fmaxf(v2, v3));
#pragma unroll
      for (int msk = 1; msk < 16; msk <<= 1) mx = fmaxf(mx, __shfl_xor(mx, msk, 32));
      float mnew = fmaxf(mrow[r], mx);
      float alpha = __expf(mrow[r] - mnew);
      float p0 = __expf(v0 - mnew), p1 = __expf(v1 - mnew);
      float p2 = __expf(v2 - mnew), p3 = __expf(v3 - mnew);
      float ps = (p0 + p1) + (p2 + p3);
#pragma unroll
      for (int msk = 1; msk < 16; msk <<= 1) ps += __shfl_xor(ps, msk, 32);
      lrow[r] = lrow[r] * alpha + ps;
      mrow[r] = mnew;
#pragma unroll
      for (int nn = 0; nn < 4; ++nn) accO[nn][r] *= alpha;
      int rg = r + half * 8;
      pw16[rg * 64 + n16]      = f2bf(p0);
      pw16[rg * 64 + 16 + n16] = f2bf(p1);
      pw16[rg * 64 + 32 + n16] = f2bf(p2);
      pw16[rg * 64 + 48 + n16] = f2bf(p3);
    }

    // P(16x64) x V(64x64): two 32-wide k-steps
#pragma unroll
    for (int ks = 0; ks < 2; ++ks) {
      Frag ap;
      ap.q[0] = pw4[n16 * 8 + ks * 4 + half];
      ap.q[1] = pw4[n16 * 8 + ks * 4 + 2 + half];
#pragma unroll
      for (int nn = 0; nn < 4; ++nn) {
        Frag bv;
        int dh = nn * 16 + n16;
        bv.q[0] = vt4[dh * 8 + ks * 4 + half];
        bv.q[1] = vt4[dh * 8 + ks * 4 + 2 + half];
        accO[nn] = __builtin_amdgcn_wmma_f32_16x16x32_bf16(
            false, frag_cast(ap), false, frag_cast(bv), (short)0, accO[nn], false, false);
      }
    }
  }

  // normalize + write ctx as bf16 [B,S,H]
#pragma unroll
  for (int r = 0; r < 8; ++r) {
    int rg = r + half * 8;
    float inv = 1.0f / lrow[r];
    size_t rowoff = ((size_t)(b * kS) + q0 + rg) * kH + h * kDH;
#pragma unroll
    for (int nn = 0; nn < 4; ++nn)
      ctx[rowoff + nn * 16 + n16] = f2bf(accO[nn][r] * inv);
  }
}

// ---------------- launch ----------------
extern "C" void kernel_launch(void* const* d_in, const int* in_sizes, int n_in,
                              void* d_out, int out_size, void* d_ws, size_t ws_size,
                              hipStream_t stream) {
  const float* latent = (const float*)d_in[0];
  const float* ln1_w = (const float*)d_in[1];
  const float* ln1_b = (const float*)d_in[2];
  const float* Wq = (const float*)d_in[3];
  const float* bq = (const float*)d_in[4];
  const float* Wk = (const float*)d_in[5];
  const float* bk = (const float*)d_in[6];
  const float* Wv = (const float*)d_in[7];
  const float* bv = (const float*)d_in[8];
  const float* Wo = (const float*)d_in[9];
  const float* bo = (const float*)d_in[10];
  const float* ln2_w = (const float*)d_in[11];
  const float* ln2_b = (const float*)d_in[12];
  const float* W1 = (const float*)d_in[13];
  const float* b1 = (const float*)d_in[14];
  const float* W2 = (const float*)d_in[15];
  const float* b2 = (const float*)d_in[16];

  char* ws = (char*)d_ws;
  size_t off = 0;
  auto alloc = [&](size_t bytes) -> void* {
    void* p = ws + off;
    off += (bytes + 255) & ~(size_t)255;
    return p;
  };
  u16* wqkv = (u16*)alloc((size_t)kH * 3 * kH * 2);
  u16* wob  = (u16*)alloc((size_t)kH * kH * 2);
  u16* w1b  = (u16*)alloc((size_t)kH * kFF * 2);
  u16* w2b  = (u16*)alloc((size_t)kFF * kH * 2);
  u16* nx   = (u16*)alloc((size_t)kM * kH * 2);
  u16* qbuf = (u16*)alloc((size_t)kM * kH * 2);
  u16* ktb  = (u16*)alloc((size_t)kM * kH * 2);
  u16* vbuf = (u16*)alloc((size_t)kM * kH * 2);
  u16* ctxb = (u16*)alloc((size_t)kM * kH * 2);
  float* res1 = (float*)alloc((size_t)kM * kH * 4);
  u16* hb   = (u16*)alloc((size_t)kM * kFF * 2);
  (void)ws_size; (void)in_sizes; (void)n_in; (void)out_size;

  // 1) weights -> bf16
  k_cvt_qkv<<<(kH * 3 * kH + 255) / 256, 256, 0, stream>>>(Wq, Wk, Wv, wqkv);
  k_cvt<<<(kH * kH + 255) / 256, 256, 0, stream>>>(Wo, wob, kH * kH);
  k_cvt<<<(kH * kFF + 255) / 256, 256, 0, stream>>>(W1, w1b, kH * kFF);
  k_cvt<<<(kFF * kH + 255) / 256, 256, 0, stream>>>(W2, w2b, kFF * kH);

  // 2) LN1
  k_ln<<<kM, 256, 0, stream>>>(latent, ln1_w, ln1_b, nx);

  // 3) fused QKV GEMM [4096 x 2304]
  k_gemm<0><<<dim3(3 * kH / 128, kM / 128), 256, 0, stream>>>(
      nx, wqkv, kH, 3 * kH, bq, bk, bv, nullptr,
      nullptr, nullptr, qbuf, ktb, vbuf);

  // 4) attention
  k_attn<<<kB * kNH * (kS / 128), 256, 0, stream>>>(qbuf, ktb, vbuf, ctxb);

  // 5) O-proj + residual -> res1 (fp32)
  k_gemm<1><<<dim3(kH / 128, kM / 128), 256, 0, stream>>>(
      ctxb, wob, kH, kH, bo, nullptr, nullptr, latent,
      res1, nullptr, nullptr, nullptr, nullptr);

  // 6) LN2
  k_ln<<<kM, 256, 0, stream>>>(res1, ln2_w, ln2_b, nx);

  // 7) FFN1 + GELU -> hb (bf16)
  k_gemm<2><<<dim3(kFF / 128, kM / 128), 256, 0, stream>>>(
      nx, w1b, kH, kFF, b1, nullptr, nullptr, nullptr,
      nullptr, hb, nullptr, nullptr, nullptr);

  // 8) FFN2 + residual -> d_out (fp32)
  k_gemm<3><<<dim3(kH / 128, kM / 128), 256, 0, stream>>>(
      hb, w2b, kFF, kH, b2, nullptr, nullptr, res1,
      (float*)d_out, nullptr, nullptr, nullptr, nullptr);
}
